// TurnEncoderAttention_81174881894436
// MI455X (gfx1250) — compile-verified, matrix operand order checked by reading, not run on hardware
//
#include <hip/hip_runtime.h>
#include <cstdint>
#include <cstddef>

// Problem constants (match reference)
constexpr int Bsz  = 4;
constexpr int Slen = 2048;
constexpr int Dm   = 1024;
constexpr int Hn   = 16;
constexpr int Hd   = 64;
constexpr int Mrows = Bsz * Slen;       // 8192
// softmax scale folded into Q at RoPE time, in base-2 units:
// (1/sqrt(64)) * log2(e)
constexpr float QSCALE = 0.125f * 1.4426950408889634f;

typedef __attribute__((ext_vector_type(16))) __bf16 v16bf;
typedef __attribute__((ext_vector_type(8)))  float  v8f;
typedef __attribute__((ext_vector_type(4)))  int    v4i;

#define AS1 __attribute__((address_space(1)))
#define AS3 __attribute__((address_space(3)))

// ---- gfx1250 async global->LDS path (compile-safe feature probe) -----------
#if defined(__has_builtin)
#  if __has_builtin(__builtin_amdgcn_global_load_async_to_lds_b128)
#    define HAVE_ASYNC_LDS 1
#  endif
#endif
#ifndef HAVE_ASYNC_LDS
#  define HAVE_ASYNC_LDS 0
#endif

// Copy 16B global -> LDS. Async (ASYNCcnt-tracked DMA) when available.
__device__ __forceinline__ void cp16(const __bf16* g, __bf16* s) {
#if HAVE_ASYNC_LDS
  __builtin_amdgcn_global_load_async_to_lds_b128(
      (AS1 v4i*)(AS1 void*)(void*)g,
      (AS3 v4i*)(AS3 void*)(void*)s,
      0, 0);
#else
  *reinterpret_cast<uint4*>(s) = *reinterpret_cast<const uint4*>(g);
#endif
}

__device__ __forceinline__ void wait_async() {
#if HAVE_ASYNC_LDS
#  if __has_builtin(__builtin_amdgcn_s_wait_asynccnt)
  __builtin_amdgcn_s_wait_asynccnt(0);
#  else
  asm volatile("s_wait_asynccnt 0" ::: "memory");
#  endif
#endif
}

union FragU { v16bf v; uint4 q[2]; };

// 16x32 bf16 WMMA fragment load (A layout; B layout is the lane-transposed
// dual with identical per-lane K sets). Lane L: row = row0 + (L&15);
// K set = k0 + 8*(L>>4) + {0..7, 16..23} -> two contiguous 16B loads.
__device__ __forceinline__ v16bf load_frag(const __bf16* base, int pitch, int row0, int k0) {
  int lane = threadIdx.x & 31;
  const __bf16* p = base + (size_t)(row0 + (lane & 15)) * pitch + k0 + ((lane >> 4) << 3);
  FragU f;
  f.q[0] = *reinterpret_cast<const uint4*>(p);
  f.q[1] = *reinterpret_cast<const uint4*>(p + 16);
  return f.v;
}

__device__ __forceinline__ v8f wmma_bf16(v16bf a, v16bf b, v8f c) {
  return __builtin_amdgcn_wmma_f32_16x16x32_bf16(false, a, false, b, (short)0, c, false, false);
}

__device__ __forceinline__ v8f vzero() {
  v8f z;
#pragma unroll
  for (int i = 0; i < 8; ++i) z[i] = 0.0f;
  return z;
}

// ---------------------------------------------------------------- casts
__global__ __launch_bounds__(256) void cast_f32_bf16(const float* __restrict__ src,
                                                     __bf16* __restrict__ dst, int n) {
  int i = blockIdx.x * 256 + threadIdx.x;
  if (i < n) dst[i] = (__bf16)src[i];
}

// w: [1024][1024] f32 row-major -> wt[n][k] bf16 (W^T), contiguous writes
__global__ __launch_bounds__(256) void transpose_cast_1024(const float* __restrict__ w,
                                                           __bf16* __restrict__ wt) {
  int t = blockIdx.x * 256 + threadIdx.x;
  int nIdx = t >> 10, kIdx = t & 1023;
  wt[(size_t)nIdx * 1024 + kIdx] = (__bf16)w[(size_t)kIdx * 1024 + nIdx];
}

// ---------------------------------------------------------------- RoPE
// q,k: bf16 [B,H,S,HD]. Q additionally pre-scaled by QSCALE so attention
// scores are already in base-2 softmax units (no per-element muls later).
__global__ __launch_bounds__(256) void rope_kernel(__bf16* __restrict__ q,
                                                   __bf16* __restrict__ k) {
  int t   = blockIdx.x * 256 + threadIdx.x;   // B*H*S*32 threads
  int i   = t & 31;
  int row = t >> 5;
  int s   = row & (Slen - 1);
  float invf = exp2f(-(float)i * (13.287712379549449f / 32.0f)); // 10000^(-2i/64)
  float ang  = (float)s * invf;
  float sn, cs;
  __sincosf(ang, &sn, &cs);
  __bf16* qp = q + (size_t)row * Hd;
  __bf16* kp = k + (size_t)row * Hd;
  float q0 = (float)qp[i], q1 = (float)qp[i + 32];
  qp[i]      = (__bf16)((q0 * cs - q1 * sn) * QSCALE);
  qp[i + 32] = (__bf16)((q1 * cs + q0 * sn) * QSCALE);
  float k0 = (float)kp[i], k1 = (float)kp[i + 32];
  kp[i]      = (__bf16)(k0 * cs - k1 * sn);
  kp[i + 32] = (__bf16)(k1 * cs + k0 * sn);
}

// ---------------------------------------------------------------- GEMM
// C[M,N] = A[M,K] (bf16) * Bt[N,K]^T (bf16 W^T) + bias
// Block tile 128x256, 8 waves of 64x64 (16 WMMAs per 8 fragment loads).
// EPI 0: f32 out [M][N]; EPI 1: bf16 [B,H,S,HD]; EPI 2: bf16 [B,H,HD,S]
template<int EPI>
__global__ __launch_bounds__(256) void gemm_bf16_kernel(
    const __bf16* __restrict__ A, const __bf16* __restrict__ Bt,
    const float* __restrict__ bias, float* __restrict__ outF,
    __bf16* __restrict__ outB, int M, int N, int K) {
  __shared__ __attribute__((aligned(16))) __bf16 sA[2][128 * 32];  // 8KB x2
  __shared__ __attribute__((aligned(16))) __bf16 sB[2][256 * 32];  // 16KB x2

  const int tid   = threadIdx.x;
  const int mBase = blockIdx.x * 128;
  const int nBase = blockIdx.y * 256;
  const int w  = tid >> 5;
  const int wm = w & 1;        // 2 M-tiles of 64 rows
  const int wn = w >> 1;       // 4 N-tiles of 64 cols

  v8f acc[4][4];
#pragma unroll
  for (int mi = 0; mi < 4; ++mi)
#pragma unroll
    for (int ni = 0; ni < 4; ++ni) acc[mi][ni] = vzero();

  auto loadTile = [&](int kt, int buf) {
    int kb = kt * 32;
    {  // A: 128 rows x 64B; 2 threads/row, 32B each
      int row = tid >> 1, half = tid & 1;
      const __bf16* ga = A + (size_t)(mBase + row) * K + kb + half * 16;
      __bf16* da = &sA[buf][row * 32 + half * 16];
      cp16(ga, da);
      cp16(ga + 8, da + 8);
    }
    {  // B: 256 rows x 64B; 1 thread/row
      int row = tid;
      const __bf16* gb = Bt + (size_t)(nBase + row) * K + kb;
      __bf16* db = &sB[buf][row * 32];
#pragma unroll
      for (int i = 0; i < 4; ++i) cp16(gb + i * 8, db + i * 8);
    }
  };

  loadTile(0, 0);
  wait_async();
  __syncthreads();
  const int nK = K >> 5;
  for (int kt = 0; kt < nK; ++kt) {
    int cur = kt & 1;
    if (kt + 1 < nK) loadTile(kt + 1, cur ^ 1);
    const __bf16* sa = sA[cur];
    const __bf16* sb = sB[cur];
    v16bf af[4], bf[4];
#pragma unroll
    for (int mi = 0; mi < 4; ++mi) af[mi] = load_frag(sa, 32, wm * 64 + mi * 16, 0);
#pragma unroll
    for (int ni = 0; ni < 4; ++ni) bf[ni] = load_frag(sb, 32, wn * 64 + ni * 16, 0);
#pragma unroll
    for (int mi = 0; mi < 4; ++mi)
#pragma unroll
      for (int ni = 0; ni < 4; ++ni)
        acc[mi][ni] = wmma_bf16(af[mi], bf[ni], acc[mi][ni]);
    wait_async();
    __syncthreads();
  }

  // Epilogue: D-fragment lane L: col = L&15, rows = r + 8*(L>>4)
  const int lane = tid & 31;
  const int nl = lane & 15, hi = lane >> 4;
#pragma unroll
  for (int mi = 0; mi < 4; ++mi) {
#pragma unroll
    for (int ni = 0; ni < 4; ++ni) {
      int gn = nBase + wn * 64 + ni * 16 + nl;
      float bv = bias[gn];
#pragma unroll
      for (int r = 0; r < 8; ++r) {
        int gm = mBase + wm * 64 + mi * 16 + r + 8 * hi;
        float v = acc[mi][ni][r] + bv;
        if (EPI == 0) {
          outF[(size_t)gm * N + gn] = v;
        } else {
          int h = gn >> 6, hd = gn & 63;
          int b = gm >> 11, s = gm & (Slen - 1);
          if (EPI == 1)
            outB[((((size_t)b * Hn + h) * Slen + s)) * Hd + hd] = (__bf16)v;
          else
            outB[((((size_t)b * Hn + h) * Hd + hd)) * Slen + s] = (__bf16)v;
        }
      }
    }
  }
}

// ---------------------------------------------------------------- attention
// q (pre-scaled by QSCALE), k: bf16 [B,H,S,HD]; vt: bf16 [B,H,HD,S];
// ctx out: bf16 [B,S,D]. 8 waves x 16 query rows; 128-key tiles double
// buffered in LDS; online softmax in registers (base-2), one butterfly
// reduction pair per 128 keys.
constexpr int KT = 128;
__global__ __launch_bounds__(256) void attn_kernel(const __bf16* __restrict__ q,
                                                   const __bf16* __restrict__ k,
                                                   const __bf16* __restrict__ vt,
                                                   __bf16* __restrict__ ctx) {
  __shared__ __attribute__((aligned(16))) __bf16 sK[2][KT * 64];   // [key][hd] 16KB x2
  __shared__ __attribute__((aligned(16))) __bf16 sV[2][64 * KT];   // [hd][key] 16KB x2
  __shared__ __attribute__((aligned(16))) __bf16 sP[8][16 * KT];   // per-wave P, 4KB each

  const int tid  = threadIdx.x;
  const int w    = tid >> 5;
  const int lane = tid & 31;
  const int nl = lane & 15, hi = lane >> 4;
  const int bh    = blockIdx.y;                 // b*H + h
  const int qBase = blockIdx.x * 128 + w * 16;

  const __bf16* qrow = q + ((size_t)bh * Slen + qBase) * Hd;
  v16bf qa0 = load_frag(qrow, Hd, 0, 0);
  v16bf qa1 = load_frag(qrow, Hd, 0, 32);

  v8f acc[4];
#pragma unroll
  for (int ni = 0; ni < 4; ++ni) acc[ni] = vzero();
  float rm[8], rl[8];
#pragma unroll
  for (int r = 0; r < 8; ++r) { rm[r] = -3.0e38f; rl[r] = 0.0f; }

  auto loadKV = [&](int tileIdx, int buf) {
    int keyBase = tileIdx * KT;
    {  // K: 128 key rows x 128B; 2 threads/row, 64B each
      int row = tid >> 1, half = tid & 1;
      const __bf16* src = k + ((size_t)bh * Slen + keyBase + row) * Hd + half * 32;
      __bf16* dst = &sK[buf][row * 64 + half * 32];
#pragma unroll
      for (int i = 0; i < 4; ++i) cp16(src + i * 8, dst + i * 8);
    }
    {  // V^T: 64 hd rows x 256B; 4 threads/row, 64B each
      int row = tid >> 2, quar = tid & 3;
      const __bf16* src = vt + ((size_t)bh * Hd + row) * Slen + keyBase + quar * 32;
      __bf16* dst = &sV[buf][row * KT + quar * 32];
#pragma unroll
      for (int i = 0; i < 4; ++i) cp16(src + i * 8, dst + i * 8);
    }
  };

  loadKV(0, 0);
  wait_async();
  __syncthreads();

  const int nTiles = Slen / KT;                 // 16
  for (int kt = 0; kt < nTiles; ++kt) {
    int cur = kt & 1;
    if (kt + 1 < nTiles) loadKV(kt + 1, cur ^ 1);
    const __bf16* skc = sK[cur];
    const __bf16* svc = sV[cur];

    // scores: 8 16x16 f32 tiles over 128 keys (K=64 -> 2 WMMAs each)
    v8f sc[8];
#pragma unroll
    for (int j = 0; j < 8; ++j) {
      sc[j] = vzero();
      sc[j] = wmma_bf16(qa0, load_frag(skc, 64, j * 16, 0),  sc[j]);
      sc[j] = wmma_bf16(qa1, load_frag(skc, 64, j * 16, 32), sc[j]);
    }

    // row max over 128 keys (8 tiles locally, then one 16-lane butterfly)
    float tmax[8];
#pragma unroll
    for (int r = 0; r < 8; ++r) {
      float t = sc[0][r];
#pragma unroll
      for (int j = 1; j < 8; ++j) t = fmaxf(t, sc[j][r]);
      tmax[r] = t;
    }
#pragma unroll
    for (int m = 1; m < 16; m <<= 1)
#pragma unroll
      for (int r = 0; r < 8; ++r) tmax[r] = fmaxf(tmax[r], __shfl_xor(tmax[r], m, 32));

    float alpha[8], psum[8];
#pragma unroll
    for (int r = 0; r < 8; ++r) {
      float mnew = fmaxf(rm[r], tmax[r]);
      alpha[r] = exp2f(rm[r] - mnew);   // base-2: scale folded into Q
      rm[r] = mnew;
      psum[r] = 0.0f;
    }

    // p = 2^(s - m), accumulate row sums, spill to per-wave LDS (A layout src)
    __bf16* pw = sP[w];
#pragma unroll
    for (int j = 0; j < 8; ++j) {
#pragma unroll
      for (int r = 0; r < 8; ++r) {
        float p = exp2f(sc[j][r] - rm[r]);
        psum[r] += p;
        pw[(r + 8 * hi) * KT + j * 16 + nl] = (__bf16)p;
      }
    }
#pragma unroll
    for (int m = 1; m < 16; m <<= 1)
#pragma unroll
      for (int r = 0; r < 8; ++r) psum[r] += __shfl_xor(psum[r], m, 32);
#pragma unroll
    for (int r = 0; r < 8; ++r) rl[r] = rl[r] * alpha[r] + psum[r];
#pragma unroll
    for (int ni = 0; ni < 4; ++ni)
#pragma unroll
      for (int r = 0; r < 8; ++r) acc[ni][r] *= alpha[r];

    asm volatile("s_wait_dscnt 0" ::: "memory");  // wave-local LDS RAW fence

    // out[16x64] += P[16x128] * V[128x64]  (16 WMMAs)
    v16bf pa[4];
#pragma unroll
    for (int kc = 0; kc < 4; ++kc) pa[kc] = load_frag(pw, KT, 0, kc * 32);
#pragma unroll
    for (int kc = 0; kc < 4; ++kc)
#pragma unroll
      for (int ni = 0; ni < 4; ++ni)
        acc[ni] = wmma_bf16(pa[kc], load_frag(svc, KT, ni * 16, kc * 32), acc[ni]);

    wait_async();
    __syncthreads();
  }

  // finalize: divide by row sums, write bf16 context [B,S,D]
  const int b = bh >> 4, h = bh & 15;
  float inv[8];
#pragma unroll
  for (int r = 0; r < 8; ++r) inv[r] = 1.0f / rl[r];
#pragma unroll
  for (int ni = 0; ni < 4; ++ni) {
#pragma unroll
    for (int r = 0; r < 8; ++r) {
      int m = r + 8 * hi;
      int s = qBase + m;
      int col = h * 64 + ni * 16 + nl;
      ctx[((size_t)b * Slen + s) * Dm + col] = (__bf16)(acc[ni][r] * inv[r]);
    }
  }
}

// ---------------------------------------------------------------- launch
extern "C" void kernel_launch(void* const* d_in, const int* in_sizes, int n_in,
                              void* d_out, int out_size, void* d_ws, size_t ws_size,
                              hipStream_t stream) {
  const float* x   = (const float*)d_in[0];
  const float* q_w = (const float*)d_in[1];
  const float* q_b = (const float*)d_in[2];
  const float* k_w = (const float*)d_in[3];
  const float* k_b = (const float*)d_in[4];
  const float* v_w = (const float*)d_in[5];
  const float* v_b = (const float*)d_in[6];
  const float* o_w = (const float*)d_in[7];
  const float* o_b = (const float*)d_in[8];
  float* out = (float*)d_out;

  __bf16* ws = (__bf16*)d_ws;
  size_t off = 0;
  __bf16* xb   = ws + off; off += (size_t)Mrows * Dm;
  __bf16* wqT  = ws + off; off += (size_t)Dm * Dm;
  __bf16* wkT  = ws + off; off += (size_t)Dm * Dm;
  __bf16* wvT  = ws + off; off += (size_t)Dm * Dm;
  __bf16* woT  = ws + off; off += (size_t)Dm * Dm;
  __bf16* qB   = ws + off; off += (size_t)Mrows * Dm;   // [B,H,S,HD]
  __bf16* kB   = ws + off; off += (size_t)Mrows * Dm;   // [B,H,S,HD]
  __bf16* vtB  = ws + off; off += (size_t)Mrows * Dm;   // [B,H,HD,S]
  __bf16* ctxB = ws + off; off += (size_t)Mrows * Dm;   // [B,S,D]

  const int nX = Mrows * Dm;
  const int nW = Dm * Dm;
  cast_f32_bf16<<<nX / 256, 256, 0, stream>>>(x, xb, nX);
  transpose_cast_1024<<<nW / 256, 256, 0, stream>>>(q_w, wqT);
  transpose_cast_1024<<<nW / 256, 256, 0, stream>>>(k_w, wkT);
  transpose_cast_1024<<<nW / 256, 256, 0, stream>>>(v_w, wvT);
  transpose_cast_1024<<<nW / 256, 256, 0, stream>>>(o_w, woT);

  dim3 gGemm(Mrows / 128, Dm / 256);          // 64 x 4
  gemm_bf16_kernel<1><<<gGemm, 256, 0, stream>>>(xb, wqT, q_b, nullptr, qB,  Mrows, Dm, Dm);
  gemm_bf16_kernel<1><<<gGemm, 256, 0, stream>>>(xb, wkT, k_b, nullptr, kB,  Mrows, Dm, Dm);
  gemm_bf16_kernel<2><<<gGemm, 256, 0, stream>>>(xb, wvT, v_b, nullptr, vtB, Mrows, Dm, Dm);

  rope_kernel<<<(Bsz * Hn * Slen * 32) / 256, 256, 0, stream>>>(qB, kB);

  attn_kernel<<<dim3(Slen / 128, Bsz * Hn), 256, 0, stream>>>(qB, kB, vtB, ctxB);

  gemm_bf16_kernel<0><<<gGemm, 256, 0, stream>>>(ctxB, woT, o_b, out, nullptr, Mrows, Dm, Dm);
}
